// RPATLite_35467839930414
// MI455X (gfx1250) — matrix operand order repaired and checked by reading, not compiled, not run on hardware
//
#include <hip/hip_runtime.h>
#include <hip/hip_bf16.h>

// ---------------------------------------------------------------------------
// RPATLite for MI455X (gfx1250): sequential RNN + GAT attention, all GEMMs via
// v_wmma_f32_16x16x32_f16 (f16 in, f32 accumulate).
//  * double-buffered LDS pipeline (1 barrier / K-step, loads overlap WMMA)
//  * A-tile staging via GLOBAL_LOAD_ASYNC_TO_LDS_B128 (ASYNCcnt path)
//  * attention probabilities generated on-the-fly into WMMA A-fragments
//    (row-max = lrelu(s_n + max_m d_m)) so NxN is never materialized.
// ---------------------------------------------------------------------------

typedef __attribute__((ext_vector_type(16))) _Float16 v16h;
typedef __attribute__((ext_vector_type(8)))  float    v8f;
typedef __attribute__((ext_vector_type(4)))  int      v4i;

static constexpr int TT = 64, NP = 2048, RR = 1024, D2 = 128, HH = 8, FF = 64;
static constexpr float ALPHA = 0.2f;

union Frag { v16h v; uint4 q[2]; _Float16 e[16]; };

#if defined(__has_builtin)
#  if __has_builtin(__builtin_amdgcn_global_load_async_to_lds_b128) && \
      __has_builtin(__builtin_amdgcn_s_wait_asynccnt)
#    define HAVE_ASYNC_LDS 1
#  endif
#endif
#ifndef HAVE_ASYNC_LDS
#  define HAVE_ASYNC_LDS 0
#endif

#if HAVE_ASYNC_LDS
typedef __attribute__((address_space(1))) v4i* g_v4i;   // global 16B vector ptr
typedef __attribute__((address_space(3))) v4i* l_v4i;   // LDS 16B vector ptr
__device__ __forceinline__ void async_a_to_lds16B(const _Float16* g, _Float16* l) {
    __builtin_amdgcn_global_load_async_to_lds_b128((g_v4i)g, (l_v4i)l, 0, 0);
}
#endif

// ---------------------------------------------------------------------------
// Generic f16 WMMA GEMM:  C[M,N] = A[M,K] @ B[K,N] (+ Cin), row-major.
// Workgroup tile 128(M) x 64(N), 8 waves, each wave 16x64 (4 accumulators).
// M % 128 == 0, N % 64 == 0, K % 32 == 0.  blockIdx.z batches (elem strides).
// Double-buffered LDS, one barrier per K-step.
// ---------------------------------------------------------------------------
__global__ __launch_bounds__(256) void wmma_gemm_kernel(
    const _Float16* __restrict__ A, const _Float16* __restrict__ B,
    const float* __restrict__ Cin, float* __restrict__ Cf32,
    _Float16* __restrict__ Cf16,
    int M, int N, int K, int lda, int ldb, int ldc,
    int sA, int sB, int sC)
{
    const int z = blockIdx.z;
    A += (size_t)z * sA;
    B += (size_t)z * sB;
    const size_t cOff = (size_t)z * sC;

    __shared__ __align__(16) _Float16 lsA[2][128 * 32];   // [row][k]  2 x 8 KB
    __shared__ __align__(16) _Float16 lsBt[2][64 * 32];   // [n][k]    2 x 4 KB

    const int t    = threadIdx.x;
    const int wave = t >> 5;
    const int lane = t & 31;
    const int r    = lane & 15;
    const int g    = lane >> 4;
    const int rowBase = blockIdx.y * 128;
    const int colBase = blockIdx.x * 64;

    // Per-thread staging coordinates.
    // A tile = 128 rows x 32 halves = 512 x 16B; thread handles items t, t+256.
    const int ar  = t >> 2;
    const int as8 = (t & 3) * 8;
    const int aL0 = ar * 32 + as8;
    const int aL1 = aL0 + 64 * 32;
    const size_t aG0 = (size_t)(rowBase + ar) * lda + as8;
    const size_t aG1 = aG0 + (size_t)64 * lda;
    // B tile = 32(k) x 64(n) halves; thread loads one 16B row chunk.
    const int bK  = t >> 3;
    const int bC8 = (t & 7) * 8;

    const v8f vzero = {0.f, 0.f, 0.f, 0.f, 0.f, 0.f, 0.f, 0.f};
    v8f acc[4] = {vzero, vzero, vzero, vzero};

    const int nk = K / 32;

    // ---- prologue: stage tile 0 into buffer 0 --------------------------------
#if HAVE_ASYNC_LDS
    async_a_to_lds16B(&A[aG0], &lsA[0][aL0]);
    async_a_to_lds16B(&A[aG1], &lsA[0][aL1]);
#else
    {
        const uint4 q0 = *(const uint4*)&A[aG0];
        const uint4 q1 = *(const uint4*)&A[aG1];
        *(uint4*)&lsA[0][aL0] = q0;
        *(uint4*)&lsA[0][aL1] = q1;
    }
#endif
    {
        const uint4 qb = *(const uint4*)&B[(size_t)bK * ldb + colBase + bC8];
        const _Float16* hv = (const _Float16*)&qb;
        #pragma unroll
        for (int j = 0; j < 8; ++j) lsBt[0][(bC8 + j) * 32 + bK] = hv[j];
    }
#if HAVE_ASYNC_LDS
    __builtin_amdgcn_s_wait_asynccnt(0);
#endif
    __syncthreads();

    // ---- main pipelined loop -------------------------------------------------
    for (int kt = 0; kt < nk; ++kt) {
        const int buf = kt & 1, nbuf = buf ^ 1;
        const bool hasNext = (kt + 1) < nk;

        uint4 qb = {};
#if !HAVE_ASYNC_LDS
        uint4 qa0 = {}, qa1 = {};
#endif
        if (hasNext) {
            const int k0n = (kt + 1) * 32;
#if HAVE_ASYNC_LDS
            async_a_to_lds16B(&A[aG0 + k0n], &lsA[nbuf][aL0]);
            async_a_to_lds16B(&A[aG1 + k0n], &lsA[nbuf][aL1]);
#else
            qa0 = *(const uint4*)&A[aG0 + k0n];
            qa1 = *(const uint4*)&A[aG1 + k0n];
#endif
            qb = *(const uint4*)&B[(size_t)(k0n + bK) * ldb + colBase + bC8];
        }

        // Compute tile kt while next tile's loads are in flight.
        Frag af;
        const int aOff = (wave * 16 + r) * 32;
        af.q[0] = *(const uint4*)&lsA[buf][aOff + 8 * g];
        af.q[1] = *(const uint4*)&lsA[buf][aOff + 16 + 8 * g];
        #pragma unroll
        for (int nt = 0; nt < 4; ++nt) {
            Frag bf;
            const int bOff = (nt * 16 + r) * 32 + 8 * g;
            bf.q[0] = *(const uint4*)&lsBt[buf][bOff];
            bf.q[1] = *(const uint4*)&lsBt[buf][bOff + 16];
            acc[nt] = __builtin_amdgcn_wmma_f32_16x16x32_f16(
                false, af.v, false, bf.v, (short)0, acc[nt], false, false);
        }

        if (hasNext) {
#if !HAVE_ASYNC_LDS
            *(uint4*)&lsA[nbuf][aL0] = qa0;
            *(uint4*)&lsA[nbuf][aL1] = qa1;
#endif
            const _Float16* hv = (const _Float16*)&qb;
            #pragma unroll
            for (int j = 0; j < 8; ++j) lsBt[nbuf][(bC8 + j) * 32 + bK] = hv[j];
#if HAVE_ASYNC_LDS
            __builtin_amdgcn_s_wait_asynccnt(0);
#endif
        }
        __syncthreads();
    }

    // Epilogue. C layout: VGPR v, lanes 0-15 -> M=v, lanes 16-31 -> M=v+8; N=r.
    const int rowOut = rowBase + wave * 16 + 8 * g;
    const int colOut = colBase + r;
    #pragma unroll
    for (int nt = 0; nt < 4; ++nt) {
        #pragma unroll
        for (int v = 0; v < 8; ++v) {
            const size_t idx = cOff + (size_t)(rowOut + v) * ldc + colOut + nt * 16;
            float c = acc[nt][v];
            if (Cin)  c += Cin[idx];
            if (Cf32) Cf32[idx] = c;
            if (Cf16) Cf16[idx] = (_Float16)c;
        }
    }
}

// ---------------------------------------------------------------------------
// Fused GAT attention: out[n, h*64+f] = sum_m softmax_m(lrelu(s_n+d_m)) Wh[h,m,f]
// A-fragment p = exp(lrelu(s_n+d_m) - lrelu(s_n+dmax)) generated in registers;
// row-sum accumulated alongside and applied post-WMMA.  Double-buffered LDS.
// ---------------------------------------------------------------------------
__global__ __launch_bounds__(256) void attn_kernel(
    const float* __restrict__ e_src, const float* __restrict__ e_dst,
    const float* __restrict__ dmax, const _Float16* __restrict__ Wh16,
    _Float16* __restrict__ attOut, int Npart)
{
    const int h    = blockIdx.z;
    const int t    = threadIdx.x;
    const int wave = t >> 5;
    const int lane = t & 31;
    const int r    = lane & 15;
    const int g    = lane >> 4;
    const int n0   = (blockIdx.x * 8 + wave) * 16;

    const float* S = e_src + (size_t)h * Npart;
    const float* D = e_dst + (size_t)h * Npart;
    const _Float16* W = Wh16 + (size_t)h * Npart * FF;

    const float dm  = dmax[h];
    const float s_n = S[n0 + r];
    const float xm  = s_n + dm;
    const float rowmax = xm > 0.f ? xm : ALPHA * xm;   // lrelu monotone => row max

    __shared__ __align__(16) _Float16 lsWt[2][64 * 32];  // Wh tile transposed [f][m]
    __shared__ float lsD[2][32];
    __shared__ float lsRS[8 * 16];

    const int wK  = t >> 3;            // staging coords for Wh tile (32 x 64)
    const int wC8 = (t & 7) * 8;

    const v8f vzero = {0.f, 0.f, 0.f, 0.f, 0.f, 0.f, 0.f, 0.f};
    v8f acc[4] = {vzero, vzero, vzero, vzero};
    float rs = 0.f;

    const int nm = Npart / 32;

    // Prologue: stage tile 0 into buffer 0.
    {
        const uint4 qw = *(const uint4*)&W[(size_t)wK * FF + wC8];
        const _Float16* hv = (const _Float16*)&qw;
        #pragma unroll
        for (int j = 0; j < 8; ++j) lsWt[0][(wC8 + j) * 32 + wK] = hv[j];
        if (t < 32) lsD[0][t] = D[t];
    }
    __syncthreads();

    for (int mt = 0; mt < nm; ++mt) {
        const int buf = mt & 1, nbuf = buf ^ 1;
        const bool hasNext = (mt + 1) < nm;

        uint4 qw = {};
        float dreg = 0.f;
        if (hasNext) {
            const int m0n = (mt + 1) * 32;
            qw = *(const uint4*)&W[(size_t)(m0n + wK) * FF + wC8];
            if (t < 32) dreg = D[m0n + t];
        }

        // Generate probability A-fragment in registers (row n0+r, ISA A layout).
        Frag pa;
        #pragma unroll
        for (int e = 0; e < 16; ++e) {
            const int kl = (e < 8) ? (8 * g + e) : (16 + 8 * g + (e - 8));
            const float x  = s_n + lsD[buf][kl];
            const float lr = x > 0.f ? x : ALPHA * x;
            const float p  = __expf(lr - rowmax);
            rs += p;
            pa.e[e] = (_Float16)p;
        }
        #pragma unroll
        for (int nt = 0; nt < 4; ++nt) {
            Frag bf;
            const int bOff = (nt * 16 + r) * 32 + 8 * g;
            bf.q[0] = *(const uint4*)&lsWt[buf][bOff];
            bf.q[1] = *(const uint4*)&lsWt[buf][bOff + 16];
            acc[nt] = __builtin_amdgcn_wmma_f32_16x16x32_f16(
                false, pa.v, false, bf.v, (short)0, acc[nt], false, false);
        }

        if (hasNext) {
            const _Float16* hv = (const _Float16*)&qw;
            #pragma unroll
            for (int j = 0; j < 8; ++j) lsWt[nbuf][(wC8 + j) * 32 + wK] = hv[j];
            if (t < 32) lsD[nbuf][t] = dreg;
        }
        __syncthreads();
    }

    // Combine the two lane groups' partial row sums (they split K mod 32).
    rs += __shfl_xor(rs, 16);
    if (g == 0) lsRS[wave * 16 + r] = rs;
    __syncthreads();

    #pragma unroll
    for (int nt = 0; nt < 4; ++nt) {
        #pragma unroll
        for (int v = 0; v < 8; ++v) {
            const int row = n0 + 8 * g + v;
            const float inv = 1.f / lsRS[wave * 16 + 8 * g + v];
            attOut[(size_t)row * (HH * FF) + h * FF + nt * 16 + r] =
                (_Float16)(acc[nt][v] * inv);
        }
    }
}

// ----------------------------- helper kernels ------------------------------
__global__ void cvt_kernel(const float* __restrict__ in, _Float16* __restrict__ out, int n) {
    const int i = blockIdx.x * blockDim.x + threadIdx.x;
    if (i < n) out[i] = (_Float16)in[i];
}

__global__ void edot_kernel(const float* __restrict__ Wh, const float* __restrict__ a_src,
                            const float* __restrict__ a_dst, float* __restrict__ e_src,
                            float* __restrict__ e_dst, int Npart) {
    const int idx = blockIdx.x * blockDim.x + threadIdx.x;      // h*Npart + n
    const int h = idx / Npart;
    const float* row = Wh + (size_t)idx * FF;
    const float* as = a_src + h * FF;
    const float* ad = a_dst + h * FF;
    float ss = 0.f, sd = 0.f;
    #pragma unroll 8
    for (int f = 0; f < FF; ++f) { ss += row[f] * as[f]; sd += row[f] * ad[f]; }
    e_src[idx] = ss;
    e_dst[idx] = sd;
}

__global__ void dmax_kernel(const float* __restrict__ e_dst, float* __restrict__ dmax, int Npart) {
    const int h = blockIdx.x;
    __shared__ float red[256];
    float m = -1e30f;
    for (int i = threadIdx.x; i < Npart; i += 256)
        m = fmaxf(m, e_dst[(size_t)h * Npart + i]);
    red[threadIdx.x] = m;
    __syncthreads();
    for (int s = 128; s > 0; s >>= 1) {
        if (threadIdx.x < s) red[threadIdx.x] = fmaxf(red[threadIdx.x], red[threadIdx.x + s]);
        __syncthreads();
    }
    if (threadIdx.x == 0) dmax[h] = red[0];
}

__global__ void final_kernel(const float* __restrict__ states_t, const float* __restrict__ y,
                             float* __restrict__ out_t, int n) {
    const int i = blockIdx.x * blockDim.x + threadIdx.x;
    if (i < n) {
        const float v = y[i];
        const float e = v > 0.f ? v : (__expf(v) - 1.f);   // ELU
        out_t[i] = states_t[i] + e;
    }
}

// ------------------------------- launcher ----------------------------------
extern "C" void kernel_launch(void* const* d_in, const int* in_sizes, int n_in,
                              void* d_out, int out_size, void* d_ws, size_t ws_size,
                              hipStream_t stream) {
    (void)in_sizes; (void)n_in; (void)out_size; (void)ws_size;
    const float* states = (const float*)d_in[0];
    const float* W_h    = (const float*)d_in[1];
    const float* W_x    = (const float*)d_in[2];
    const float* W_y    = (const float*)d_in[3];
    const float* h_0    = (const float*)d_in[4];
    const float* pat_W  = (const float*)d_in[5];
    const float* a_src  = (const float*)d_in[6];
    const float* a_dst  = (const float*)d_in[7];
    const float* W_out  = (const float*)d_in[8];
    float* out = (float*)d_out;

    char* ws = (char*)d_ws;
    size_t off = 0;
    auto carve = [&](size_t bytes) -> void* {
        off = (off + 255) & ~(size_t)255;
        void* p = ws + off;
        off += bytes;
        return p;
    };
    _Float16* Whf16      = (_Float16*)carve((size_t)RR * RR * 2);
    _Float16* Wxf16      = (_Float16*)carve((size_t)RR * NP * 2);
    _Float16* Wyf16      = (_Float16*)carve((size_t)NP * RR * 2);
    _Float16* patWf16    = (_Float16*)carve((size_t)HH * D2 * FF * 2);
    _Float16* patWoutf16 = (_Float16*)carve((size_t)HH * FF * D2 * 2);
    _Float16* hf16       = (_Float16*)carve((size_t)RR * D2 * 2);
    float*    hacc       = (float*)   carve((size_t)RR * D2 * 4);
    _Float16* xtf16      = (_Float16*)carve((size_t)NP * D2 * 2);
    _Float16* xwf16      = (_Float16*)carve((size_t)NP * D2 * 2);
    float*    Whf32b     = (float*)   carve((size_t)HH * NP * FF * 4);
    _Float16* Whf16b     = (_Float16*)carve((size_t)HH * NP * FF * 2);
    float*    esrc       = (float*)   carve((size_t)HH * NP * 4);
    float*    edst       = (float*)   carve((size_t)HH * NP * 4);
    float*    dmaxb      = (float*)   carve((size_t)HH * 4);
    _Float16* attf16     = (_Float16*)carve((size_t)NP * HH * FF * 2);
    float*    yacc       = (float*)   carve((size_t)NP * D2 * 4);

    auto cvt = [&](const float* src, _Float16* dst, int n) {
        cvt_kernel<<<(n + 255) / 256, 256, 0, stream>>>(src, dst, n);
    };
    // One-time f32 -> f16 weight conversion (deterministic each call).
    cvt(W_h,   Whf16,      RR * RR);
    cvt(W_x,   Wxf16,      RR * NP);
    cvt(W_y,   Wyf16,      NP * RR);
    cvt(pat_W, patWf16,    HH * D2 * FF);
    cvt(W_out, patWoutf16, HH * FF * D2);
    cvt(h_0,   hf16,       RR * D2);

    for (int t = 0; t < TT; ++t) {
        const float* xt = states + (size_t)t * NP * D2;
        cvt(xt, xtf16, NP * D2);

        // h_acc = W_h @ h_prev                 [1024x1024]@[1024x128]
        wmma_gemm_kernel<<<dim3(D2 / 64, RR / 128, 1), 256, 0, stream>>>(
            Whf16, hf16, nullptr, hacc, nullptr,
            RR, D2, RR, RR, D2, D2, 0, 0, 0);
        // h = h_acc + W_x @ x_t  (write f16)   [1024x2048]@[2048x128]
        wmma_gemm_kernel<<<dim3(D2 / 64, RR / 128, 1), 256, 0, stream>>>(
            Wxf16, xtf16, hacc, nullptr, hf16,
            RR, D2, NP, NP, D2, D2, 0, 0, 0);
        // xw = W_y @ h                         [2048x1024]@[1024x128]
        wmma_gemm_kernel<<<dim3(D2 / 64, NP / 128, 1), 256, 0, stream>>>(
            Wyf16, hf16, nullptr, nullptr, xwf16,
            NP, D2, RR, RR, D2, D2, 0, 0, 0);
        // Wh[h] = xw @ pat_W[h]   (batched over 8 heads)
        wmma_gemm_kernel<<<dim3(FF / 64, NP / 128, HH), 256, 0, stream>>>(
            xwf16, patWf16, nullptr, Whf32b, Whf16b,
            NP, FF, D2, D2, FF, FF, 0, D2 * FF, NP * FF);
        // e_src / e_dst dots and per-head max(e_dst)
        edot_kernel<<<(HH * NP) / 256, 256, 0, stream>>>(Whf32b, a_src, a_dst,
                                                         esrc, edst, NP);
        dmax_kernel<<<HH, 256, 0, stream>>>(edst, dmaxb, NP);
        // Fused softmax-attention @ Wh -> att[2048, 512] f16
        attn_kernel<<<dim3(NP / 128, 1, HH), 256, 0, stream>>>(
            esrc, edst, dmaxb, Whf16b, attf16, NP);
        // yacc = att @ W_out                   [2048x512]@[512x128]
        wmma_gemm_kernel<<<dim3(D2 / 64, NP / 128, 1), 256, 0, stream>>>(
            attf16, patWoutf16, nullptr, yacc, nullptr,
            NP, D2, HH * FF, HH * FF, D2, D2, 0, 0, 0);
        // out[t] = states[t] + elu(yacc)
        final_kernel<<<(NP * D2) / 256, 256, 0, stream>>>(
            xt, yacc, out + (size_t)t * NP * D2, NP * D2);
    }
}
